// Linear_36034775613924
// MI455X (gfx1250) — compile-verified, matrix operand order checked by reading, not run on hardware
//
#include <hip/hip_runtime.h>
#include <stdint.h>

// ---------------------------------------------------------------------------
// Meta-linear (B=128, D=512, ORDER=1, DEPTH=2) for gfx1250 / MI455X.
// x_new[b,o] = (1/sqrt(D)) * sum_{i,k} W[o,i,k] x[b,i] x[b,k] + bias[o]
// Implemented as tall-K GEMM with per-i column scaling, bf16 hi/lo (bf16x3)
// WMMA emulation of fp32, LDS staging + async-to-LDS, deterministic K-split.
// ---------------------------------------------------------------------------

typedef __attribute__((ext_vector_type(16))) __bf16 v16bf;
typedef __attribute__((ext_vector_type(8)))  float  v8f;

#define DDIM 512
#define BDIM 128
#define KSPLIT 64          // 64 K-splits -> 8 'i' values per workgroup

// workspace byte offsets
#define WS_XFH   0u        // X fragments hi, bf16, frag layout: 128 KB
#define WS_XFL   131072u   // X fragments lo: 128 KB (must be contiguous after hi)
#define WS_XT    262144u   // x transposed f32 [i][b]: 256 KB
#define WS_X1    524288u   // depth-0 output x1 f32 [b][o]: 256 KB
#define WS_PART  786432u   // partials f32 [ks][o][b]: 16 MB

__device__ __forceinline__ void split_pack2(float a0, float a1,
                                            uint32_t& hi, uint32_t& lo) {
  __bf16 h0 = (__bf16)a0, h1 = (__bf16)a1;
  float r0 = a0 - (float)h0, r1 = a1 - (float)h1;
  __bf16 l0 = (__bf16)r0, l1 = (__bf16)r1;
  hi = (uint32_t)__builtin_bit_cast(unsigned short, h0) |
       ((uint32_t)__builtin_bit_cast(unsigned short, h1) << 16);
  lo = (uint32_t)__builtin_bit_cast(unsigned short, l0) |
       ((uint32_t)__builtin_bit_cast(unsigned short, l1) << 16);
}

union Frag {
  uint4 q[2];
  v16bf v;
};

// ---------------------------------------------------------------------------
// Pre-split x into bf16 hi/lo in WMMA-B fragment order.
// Fragment layout (mirrors 16-bit A layout over N instead of M):
//   lane l: n = l&15, kb = (l>>4)*8 ; pair p<4 -> k = c16*32+kb+2p(+1)
//                                     pair p>=4 -> k = c16*32+16+kb+2(p-4)(+1)
// Index: ((c16*8 + ntile)*32 + lane)*8 + p    (u32 = packed bf16 pair)
// ---------------------------------------------------------------------------
__global__ void prep_frag(const float* __restrict__ xin,
                          uint32_t* __restrict__ xfh,
                          uint32_t* __restrict__ xfl) {
  int t = blockIdx.x * 256 + threadIdx.x;       // 0..4095
  int c16  = t >> 8;                            // 0..15 (k chunk of 32)
  int lane = t & 31;
  int nt   = (t >> 5) & 7;                      // n-tile 0..7
  int b    = nt * 16 + (lane & 15);
  int kb   = (lane >> 4) * 8;
  const float* xr = xin + b * DDIM + c16 * 32 + kb;
  uint32_t* ph = xfh + t * 8;
  uint32_t* pl = xfl + t * 8;
#pragma unroll
  for (int p = 0; p < 4; ++p) {
    uint32_t h, l;
    split_pack2(xr[2 * p], xr[2 * p + 1], h, l);
    ph[p] = h; pl[p] = l;
    split_pack2(xr[16 + 2 * p], xr[16 + 2 * p + 1], h, l);
    ph[p + 4] = h; pl[p + 4] = l;
  }
}

// xT[i][b] = x[b][i]  (f32, for the per-i column scale)
__global__ void prep_T(const float* __restrict__ xin, float* __restrict__ xT) {
  int t = blockIdx.x * 256 + threadIdx.x;       // 65536
  int i = t >> 7, b = t & 127;
  xT[t] = xin[b * DDIM + i];
}

// ---------------------------------------------------------------------------
// Main GEMM. Grid: 256 WGs = 4 o-blocks(128) x 64 K-splits. Block: 256 thr.
// Wave tile 64(o) x 32(b); WG tile 128(o) x 128(b).
// LDS: [0,256K) X fragments (hi|lo) via async copy; [256K,288K) W hi/lo stage.
// ---------------------------------------------------------------------------
__global__ __launch_bounds__(256, 1)
void metalinear_gemm(const float* __restrict__ W,
                     const uint32_t* __restrict__ xfrag,   // hi||lo, 256 KB
                     const float* __restrict__ xT,
                     float* __restrict__ partial) {
  extern __shared__ char smem[];
  uint32_t* ldsX  = (uint32_t*)smem;                 // hi [0,32768) lo [32768,65536) u32
  uint32_t* ldsWh = (uint32_t*)(smem + 262144);      // 4096 u32
  uint32_t* ldsWl = (uint32_t*)(smem + 278528);      // 4096 u32

  const int tid    = threadIdx.x;
  const int lane   = tid & 31;
  const int wave   = tid >> 5;
  const int ob     = blockIdx.x >> 6;                // o-block of 128
  const int ks     = blockIdx.x & 63;                // K-split
  const int ostrip = wave >> 2;                      // 0..1
  const int bstrip = wave & 3;                       // 0..3

  // ---- bulk async copy of X fragments (hi+lo = 256 KB) into LDS ----
  {
    const uint64_t gbase = (uint64_t)(uintptr_t)xfrag;
    for (int j = 0; j < 64; ++j) {
      uint32_t off  = (uint32_t)(j * 256 + tid) * 16u;
      uint32_t ldsa = (uint32_t)(uintptr_t)(smem + off);   // low 32 bits = LDS offset
      uint64_t ga   = gbase + off;
      asm volatile("global_load_async_to_lds_b128 %0, %1, off"
                   :: "v"(ldsa), "v"(ga) : "memory");
    }
    asm volatile("s_wait_asynccnt 0x0" ::: "memory");
  }
  __syncthreads();

  v8f xacc[4][2];
#pragma unroll
  for (int ot = 0; ot < 4; ++ot)
#pragma unroll
    for (int nt = 0; nt < 2; ++nt)
      xacc[ot][nt] = (v8f){0.f, 0.f, 0.f, 0.f, 0.f, 0.f, 0.f, 0.f};

  for (int ii = 0; ii < 8; ++ii) {
    const int i = ks * 8 + ii;                       // contraction 'i' index
    v8f acc[4][2];
#pragma unroll
    for (int ot = 0; ot < 4; ++ot)
#pragma unroll
      for (int nt = 0; nt < 2; ++nt)
        acc[ot][nt] = (v8f){0.f, 0.f, 0.f, 0.f, 0.f, 0.f, 0.f, 0.f};

    for (int s = 0; s < 8; ++s) {                    // 8 stages x 2 chunks = K 512
      __syncthreads();
      // ---- cooperative W stage: fp32 -> bf16 hi/lo in A-fragment order ----
#pragma unroll
      for (int it = 0; it < 2; ++it) {
        int item = tid + it * 256;                   // 0..511 = (c, ot, lane)
        int c  = item >> 8;
        int ot = (item >> 5) & 7;
        int l  = item & 31;
        int o  = ob * 128 + ot * 16 + (l & 15);
        int kb = (l >> 4) * 8;
        const float* wr = W + (size_t)o * (DDIM * DDIM) + (size_t)i * DDIM
                            + s * 64 + c * 32 + kb;
        float4 a0 = *(const float4*)(wr);
        float4 a1 = *(const float4*)(wr + 4);
        float4 a2 = *(const float4*)(wr + 16);
        float4 a3 = *(const float4*)(wr + 20);
        __builtin_prefetch(wr + 64, 0, 1);           // next stage / next i row
        uint32_t h[8], lo_[8];
        split_pack2(a0.x, a0.y, h[0], lo_[0]);
        split_pack2(a0.z, a0.w, h[1], lo_[1]);
        split_pack2(a1.x, a1.y, h[2], lo_[2]);
        split_pack2(a1.z, a1.w, h[3], lo_[3]);
        split_pack2(a2.x, a2.y, h[4], lo_[4]);
        split_pack2(a2.z, a2.w, h[5], lo_[5]);
        split_pack2(a3.x, a3.y, h[6], lo_[6]);
        split_pack2(a3.z, a3.w, h[7], lo_[7]);
        uint32_t base = (uint32_t)((c * 8 + ot) * 32 + l) * 8;
        uint4* dh = (uint4*)(ldsWh + base);
        uint4* dl = (uint4*)(ldsWl + base);
        dh[0] = make_uint4(h[0], h[1], h[2], h[3]);
        dh[1] = make_uint4(h[4], h[5], h[6], h[7]);
        dl[0] = make_uint4(lo_[0], lo_[1], lo_[2], lo_[3]);
        dl[1] = make_uint4(lo_[4], lo_[5], lo_[6], lo_[7]);
      }
      __syncthreads();

#pragma unroll
      for (int c = 0; c < 2; ++c) {
        const int c16 = s * 2 + c;
        Frag bh[2], bl[2];
#pragma unroll
        for (int nt = 0; nt < 2; ++nt) {
          int ntg = bstrip * 2 + nt;
          uint32_t bidx = (uint32_t)((c16 * 8 + ntg) * 32 + lane) * 8;
          const uint4* ph = (const uint4*)(ldsX + bidx);
          const uint4* pl = (const uint4*)(ldsX + 32768 + bidx);
          bh[nt].q[0] = ph[0]; bh[nt].q[1] = ph[1];
          bl[nt].q[0] = pl[0]; bl[nt].q[1] = pl[1];
        }
#pragma unroll
        for (int ot = 0; ot < 4; ++ot) {
          int otg = ostrip * 4 + ot;
          uint32_t aidx = (uint32_t)((c * 8 + otg) * 32 + lane) * 8;
          Frag ah, al;
          const uint4* pwh = (const uint4*)(ldsWh + aidx);
          const uint4* pwl = (const uint4*)(ldsWl + aidx);
          ah.q[0] = pwh[0]; ah.q[1] = pwh[1];
          al.q[0] = pwl[0]; al.q[1] = pwl[1];
#pragma unroll
          for (int nt = 0; nt < 2; ++nt) {
            // fp32-ish via bf16x3: Ah*Bh + Ah*Bl + Al*Bh
            acc[ot][nt] = __builtin_amdgcn_wmma_f32_16x16x32_bf16(
                false, ah.v, false, bh[nt].v, (short)0, acc[ot][nt], false, false);
            acc[ot][nt] = __builtin_amdgcn_wmma_f32_16x16x32_bf16(
                false, ah.v, false, bl[nt].v, (short)0, acc[ot][nt], false, false);
            acc[ot][nt] = __builtin_amdgcn_wmma_f32_16x16x32_bf16(
                false, al.v, false, bh[nt].v, (short)0, acc[ot][nt], false, false);
          }
        }
      }
    }

    // ---- per-i column scale: xacc += acc * x[b,i]  (f32, per-lane scalar) ----
#pragma unroll
    for (int nt = 0; nt < 2; ++nt) {
      float sI = xT[i * 128 + bstrip * 32 + nt * 16 + (lane & 15)];
#pragma unroll
      for (int ot = 0; ot < 4; ++ot)
#pragma unroll
        for (int r = 0; r < 8; ++r)
          xacc[ot][nt][r] += acc[ot][nt][r] * sI;
    }
  }

  // ---- write deterministic partials: partial[ks][o][b] ----
#pragma unroll
  for (int ot = 0; ot < 4; ++ot)
#pragma unroll
    for (int nt = 0; nt < 2; ++nt)
#pragma unroll
      for (int r = 0; r < 8; ++r) {
        int o = ob * 128 + (ostrip * 4 + ot) * 16 + r + 8 * (lane >> 4);
        int b = bstrip * 32 + nt * 16 + (lane & 15);
        partial[(((size_t)ks * 512 + o) << 7) + b] = xacc[ot][nt][r];
      }
}

// xout[b][o] = (1/sqrt(D)) * sum_ks partial[ks][o][b] + bias[o]
__global__ void finalize(const float* __restrict__ partial,
                         const float* __restrict__ bias,
                         float* __restrict__ xout) {
  int t = blockIdx.x * 256 + threadIdx.x;       // 65536 = b*512 + o
  int b = t >> 9, o = t & 511;
  float s = 0.f;
#pragma unroll 8
  for (int ks = 0; ks < KSPLIT; ++ks)
    s += partial[(((size_t)ks * 512 + o) << 7) + b];
  xout[t] = s * 0.04419417382415922f + bias[o];  // 1/sqrt(512)
}

extern "C" void kernel_launch(void* const* d_in, const int* in_sizes, int n_in,
                              void* d_out, int out_size, void* d_ws, size_t ws_size,
                              hipStream_t stream) {
  const float* x0   = (const float*)d_in[0];     // (128, 512) f32
  const float* W    = (const float*)d_in[1];     // (512, 512, 512) f32
  const float* bias = (const float*)d_in[2];     // (512,) f32

  char* ws = (char*)d_ws;
  uint32_t* xfh  = (uint32_t*)(ws + WS_XFH);
  uint32_t* xfl  = (uint32_t*)(ws + WS_XFL);
  float* xT      = (float*)(ws + WS_XT);
  float* x1      = (float*)(ws + WS_X1);
  float* partial = (float*)(ws + WS_PART);

  const float* xin = x0;
  for (int depth = 0; depth < 2; ++depth) {
    prep_frag<<<16, 256, 0, stream>>>(xin, xfh, xfl);
    prep_T<<<256, 256, 0, stream>>>(xin, xT);
    metalinear_gemm<<<256, 256, 294912, stream>>>(W, xfh, xT, partial);
    finalize<<<256, 256, 0, stream>>>(partial, bias,
                                      depth == 0 ? x1 : (float*)d_out);
    xin = x1;
  }
  (void)in_sizes; (void)n_in; (void)out_size; (void)ws_size;
}
